// Seq2Seq_79980880986271
// MI455X (gfx1250) — matrix operand order repaired and checked by reading, not compiled, not run on hardware
//
#include <hip/hip_runtime.h>
#include <math.h>

// ---------------------------------------------------------------------------
// Problem constants (from reference)
// ---------------------------------------------------------------------------
#define B_    64
#define S_IN  64
#define T_TAR 32
#define V_    32000
#define E_    256
#define H_    512
#define H3_   1536
#define BEGIN_TOK 1

typedef __attribute__((ext_vector_type(16))) __bf16 v16bf;
typedef __attribute__((ext_vector_type(8)))  float  v8f;

union Frag16 { v16bf v; unsigned short u[16]; uint4 q[2]; };

__device__ __forceinline__ unsigned short f32_to_bf16(float f) {
    unsigned int u = __float_as_uint(f);
    unsigned int r = (u + 0x7FFFu + ((u >> 16) & 1u)) >> 16;   // RNE
    return (unsigned short)r;
}
__device__ __forceinline__ unsigned int pack_bf16x2(float a, float b) {
    return (unsigned int)f32_to_bf16(a) | ((unsigned int)f32_to_bf16(b) << 16);
}
__device__ __forceinline__ float sigm(float x) { return 1.0f / (1.0f + expf(-x)); }

// Async global -> LDS 16B copy (CDNA5 GLOBAL_LOAD_ASYNC_TO_LDS_B128, ASYNCcnt).
// LDS offset = low 32 bits of the flat address (ISA: LDS aperture tag is
// entirely in addr[63:32]; addr[31:0] is the LDS byte offset).
__device__ __forceinline__ void async_b128(void* lds_ptr, const void* gptr) {
    unsigned loff = (unsigned)(size_t)lds_ptr;
    asm volatile("global_load_async_to_lds_b128 %0, %1, off"
                 :: "v"(loff), "v"(gptr) : "memory");
}
#define WAIT_ASYNC(n) asm volatile("s_wait_asynccnt %0" :: "i"(n) : "memory")

// ---------------------------------------------------------------------------
// fp32 -> bf16 with transpose: src [K, N] f32 row-major -> dst [N, K] bf16.
// Tiled through LDS so both global accesses are coalesced.
// ---------------------------------------------------------------------------
#define TT 32
__global__ __launch_bounds__(256)
void cvt_transpose_bf16(const float* __restrict__ src,
                        unsigned short* __restrict__ dst, int K, int N) {
    __shared__ float tile[TT][TT + 1];
    int kt = blockIdx.y * TT, nt = blockIdx.x * TT;
    for (int i = threadIdx.y; i < TT; i += blockDim.y)
        tile[i][threadIdx.x] = src[(size_t)(kt + i) * N + nt + threadIdx.x];
    __syncthreads();
    for (int i = threadIdx.y; i < TT; i += blockDim.y)
        dst[(size_t)(nt + i) * K + kt + threadIdx.x] =
            f32_to_bf16(tile[threadIdx.x][i]);
}

// ---------------------------------------------------------------------------
// Embedding gather -> bf16 activations  (block = one batch row, 64 threads)
// ---------------------------------------------------------------------------
__global__ __launch_bounds__(64)
void gather_embed_bf16(const float* __restrict__ emb,
                       const int* __restrict__ toks,
                       int tok_stride, int tok_off,
                       unsigned short* __restrict__ Xb, int E) {
    int b = blockIdx.x;
    int t = toks[b * tok_stride + tok_off];
    const float4* row = (const float4*)(emb + (size_t)t * E);
    uint2* dst = (uint2*)(Xb + (size_t)b * E);
    for (int e = threadIdx.x; e < (E >> 2); e += blockDim.x) {
        float4 f = row[e];
        dst[e] = make_uint2(pack_bf16x2(f.x, f.y), pack_bf16x2(f.z, f.w));
    }
}

// ---------------------------------------------------------------------------
// bf16 WMMA GEMM:  C[64, ldc](+bias) = A[64, K] @ W   (64-col block tile)
// W pre-transposed: Bt [N, K] bf16 row-major, so A and B panels are contiguous
// and streamed with async global->LDS b128, double-buffered (ASYNCcnt).
// Block = 128 threads = 4 waves. Wave w owns N-subtile w (16 cols) and loops
// over the 4 M-tiles with 4 f32 accumulators, reusing its B fragment.
// Per K-stage: 2 kk-slices x (1 B frag + 4 A frags + 4 WMMAs).
// Fragment layouts per CDNA5 ISA 7.12.2:
//   A 16x32 bf16: lane row=M; elems 0..7 = K khalf..+7 (khalf=0/8 by lane
//                 half), elems 8..15 = K khalf+16..+23
//   B 32x16 bf16: lane col=N=lane%16; elems 0..7 = K kbb..+7 (kbb=0/8 by
//                 lane half), elems 8..15 = K kbb+16..+23
// ---------------------------------------------------------------------------
#define KSTEP 64
#define NBLK  64
__global__ __launch_bounds__(128)
void wmma_gemm_bf16(const unsigned short* __restrict__ A, int lda,
                    const unsigned short* __restrict__ Bt, int ldbt,
                    const float* __restrict__ bias,
                    float* __restrict__ C, int ldc, int K) {
    __shared__ unsigned short sA[2][64][KSTEP];     // 2 x 8 KB
    __shared__ unsigned short sB[2][NBLK][KSTEP];   // 2 x 8 KB

    const int tid   = threadIdx.x;
    const int lane  = tid & 31;
    const int wave  = tid >> 5;             // N-subtile 0..3
    const int n0    = blockIdx.x * NBLK;
    const int row16 = lane & 15;
    const bool hi   = lane >= 16;
    const int khalf = hi ? 8 : 0;           // A fragment K split
    const int kbb   = hi ? 8 : 0;           // B fragment K split

    // each thread issues exactly 8 async b128 per stage (4 for A, 4 for B)
    auto issue_stage = [&](int k0, int buf) {
        #pragma unroll
        for (int i = 0; i < 4; ++i) {
            int idx = tid + i * 128;        // 0..511 : A = 64 rows x 8 chunks
            int r = idx >> 3, c = idx & 7;
            async_b128(&sA[buf][r][c * 8], A + (size_t)r * lda + k0 + c * 8);
        }
        #pragma unroll
        for (int i = 0; i < 4; ++i) {
            int idx = tid + i * 128;        // 0..511 : B = 64 rows x 8 chunks
            int n = idx >> 3, c = idx & 7;
            async_b128(&sB[buf][n][c * 8],
                       Bt + (size_t)(n0 + n) * ldbt + k0 + c * 8);
        }
    };

    v8f acc0 = {}, acc1 = {}, acc2 = {}, acc3 = {};
    issue_stage(0, 0);

    int s = 0;
    for (int k0 = 0; k0 < K; k0 += KSTEP, s ^= 1) {
        if (k0 + KSTEP < K) {
            issue_stage(k0 + KSTEP, s ^ 1);   // stream next panel
            WAIT_ASYNC(8);                    // current panel landed
        } else {
            WAIT_ASYNC(0);
        }
        __syncthreads();

        const unsigned short* bRow = &sB[s][wave * 16 + row16][0];
        #pragma unroll
        for (int kk = 0; kk < KSTEP; kk += 32) {
            Frag16 fb;
            fb.q[0] = *(const uint4*)(bRow + kk + kbb);
            fb.q[1] = *(const uint4*)(bRow + kk + kbb + 16);
            Frag16 fa;
            fa.q[0] = *(const uint4*)(&sA[s][ 0 + row16][kk + khalf]);
            fa.q[1] = *(const uint4*)(&sA[s][ 0 + row16][kk + khalf + 16]);
            acc0 = __builtin_amdgcn_wmma_f32_16x16x32_bf16(
                false, fa.v, false, fb.v, (short)0, acc0, false, false);
            fa.q[0] = *(const uint4*)(&sA[s][16 + row16][kk + khalf]);
            fa.q[1] = *(const uint4*)(&sA[s][16 + row16][kk + khalf + 16]);
            acc1 = __builtin_amdgcn_wmma_f32_16x16x32_bf16(
                false, fa.v, false, fb.v, (short)0, acc1, false, false);
            fa.q[0] = *(const uint4*)(&sA[s][32 + row16][kk + khalf]);
            fa.q[1] = *(const uint4*)(&sA[s][32 + row16][kk + khalf + 16]);
            acc2 = __builtin_amdgcn_wmma_f32_16x16x32_bf16(
                false, fa.v, false, fb.v, (short)0, acc2, false, false);
            fa.q[0] = *(const uint4*)(&sA[s][48 + row16][kk + khalf]);
            fa.q[1] = *(const uint4*)(&sA[s][48 + row16][kk + khalf + 16]);
            acc3 = __builtin_amdgcn_wmma_f32_16x16x32_bf16(
                false, fa.v, false, fb.v, (short)0, acc3, false, false);
        }
        __syncthreads();
    }

    const int ncol = n0 + wave * 16 + row16;
    float bval = bias ? bias[ncol] : 0.0f;
    // C/D layout: VGPR v -> row v (lanes 0-15) / v+8 (lanes 16-31), N=lane%16
    #pragma unroll
    for (int v = 0; v < 8; ++v) {
        int mr = v + (hi ? 8 : 0);
        C[(size_t)(mr     ) * ldc + ncol] = acc0[v] + bval;
        C[(size_t)(mr + 16) * ldc + ncol] = acc1[v] + bval;
        C[(size_t)(mr + 32) * ldc + ncol] = acc2[v] + bval;
        C[(size_t)(mr + 48) * ldc + ncol] = acc3[v] + bval;
    }
}

// ---------------------------------------------------------------------------
// GRU gate kernel 1: z = sig(xz+hz), r = sig(xr+hr); emit z (f32), r*h (bf16)
// ---------------------------------------------------------------------------
__global__ __launch_bounds__(512)
void gru_zr(const float* __restrict__ G,   // [64,1536] x@Wx + b
            const float* __restrict__ G2,  // [64,1536] h@Wh
            const float* __restrict__ H,   // [64,512]
            float* __restrict__ Z, unsigned short* __restrict__ RHb) {
    int b = blockIdx.x, j = threadIdx.x;
    float z = sigm(G[b * H3_ + j]      + G2[b * H3_ + j]);
    float r = sigm(G[b * H3_ + H_ + j] + G2[b * H3_ + H_ + j]);
    Z[b * H_ + j]   = z;
    RHb[b * H_ + j] = f32_to_bf16(r * H[b * H_ + j]);
}

// ---------------------------------------------------------------------------
// GRU gate kernel 2: hh = tanh(xh + (r*h)@Whh); h = z*h + (1-z)*hh
// ---------------------------------------------------------------------------
__global__ __launch_bounds__(512)
void gru_hnew(const float* __restrict__ G,   // [64,1536]
              const float* __restrict__ M3,  // [64,512] (r*h)@Whh
              const float* __restrict__ Z,
              float* __restrict__ H, unsigned short* __restrict__ Hb) {
    int b = blockIdx.x, j = threadIdx.x;
    float hh = tanhf(G[b * H3_ + 2 * H_ + j] + M3[b * H_ + j]);
    float z  = Z[b * H_ + j];
    float hn = z * H[b * H_ + j] + (1.0f - z) * hh;
    H[b * H_ + j]  = hn;
    Hb[b * H_ + j] = f32_to_bf16(hn);
}

// ---------------------------------------------------------------------------
// Decoder head: p = softmax(L); ce = log(sum_j exp(p_j)) - p[targ] (faithful
// double softmax); tok_next = argmax L.  One block per batch row.
// ---------------------------------------------------------------------------
__global__ __launch_bounds__(256)
void dec_softmax_loss(const float* __restrict__ logits,
                      const int* __restrict__ targ, int tstep,
                      int* __restrict__ tok, float* __restrict__ ce_buf) {
    int b = blockIdx.x, tid = threadIdx.x;
    const float* L = logits + (size_t)b * V_;
    __shared__ float sf[256];
    __shared__ int   si[256];

    // pass 1: max + argmax (first-max tie-break like jnp.argmax)
    float m = -3.4e38f; int am = V_;
    for (int i = tid; i < V_; i += 256) {
        float v = L[i];
        if (v > m) { m = v; am = i; }
    }
    sf[tid] = m; si[tid] = am; __syncthreads();
    for (int s = 128; s > 0; s >>= 1) {
        if (tid < s) {
            if (sf[tid + s] > sf[tid] ||
                (sf[tid + s] == sf[tid] && si[tid + s] < si[tid])) {
                sf[tid] = sf[tid + s]; si[tid] = si[tid + s];
            }
        }
        __syncthreads();
    }
    m = sf[0]; am = si[0]; __syncthreads();

    // pass 2: softmax denominator
    float se = 0.0f;
    for (int i = tid; i < V_; i += 256) se += expf(L[i] - m);
    sf[tid] = se; __syncthreads();
    for (int s = 128; s > 0; s >>= 1) {
        if (tid < s) sf[tid] += sf[tid + s];
        __syncthreads();
    }
    float denom = sf[0]; __syncthreads();

    // pass 3: sum of exp(p_i) for the double-softmax logsumexp (p_i <= 1)
    float sep = 0.0f;
    for (int i = tid; i < V_; i += 256) sep += expf(expf(L[i] - m) / denom);
    sf[tid] = sep; __syncthreads();
    for (int s = 128; s > 0; s >>= 1) {
        if (tid < s) sf[tid] += sf[tid + s];
        __syncthreads();
    }

    if (tid == 0) {
        int tgt    = targ[b * T_TAR + tstep];
        float p_t  = expf(L[tgt] - m) / denom;
        float ce   = logf(sf[0]) - p_t;
        float mask = (tgt != 0) ? 1.0f : 0.0f;
        ce_buf[b]  = ce * mask;
        tok[b]     = am;
    }
}

__global__ __launch_bounds__(64)
void accum_loss(const float* __restrict__ ce, float* __restrict__ loss) {
    __shared__ float s[64];
    int t = threadIdx.x;
    s[t] = ce[t]; __syncthreads();
    for (int k = 32; k > 0; k >>= 1) {
        if (t < k) s[t] += s[t + k];
        __syncthreads();
    }
    if (t == 0) loss[0] += s[0] / (float)B_;
}

__global__ void init_state(float* __restrict__ H, unsigned short* __restrict__ Hb,
                           int* __restrict__ tok, float* __restrict__ loss) {
    int i = blockIdx.x * blockDim.x + threadIdx.x;
    if (i < B_ * H_) { H[i] = 0.0f; Hb[i] = 0; }
    if (i < B_)       tok[i] = BEGIN_TOK;
    if (i == 0)       loss[0] = 0.0f;
}

__global__ void write_out(const float* __restrict__ loss, float* __restrict__ out) {
    out[0] = loss[0];
}

// ---------------------------------------------------------------------------
// Host-side orchestration
// ---------------------------------------------------------------------------
extern "C" void kernel_launch(void* const* d_in, const int* in_sizes, int n_in,
                              void* d_out, int out_size, void* d_ws, size_t ws_size,
                              hipStream_t stream) {
    const int*   inp     = (const int*)  d_in[0];   // [64,64]
    const int*   targ    = (const int*)  d_in[1];   // [64,32]
    const float* emb_enc = (const float*)d_in[2];   // [32000,256]
    const float* enc_Wx  = (const float*)d_in[3];   // [256,1536]
    const float* enc_Wh  = (const float*)d_in[4];   // [512,1536]
    const float* enc_b   = (const float*)d_in[5];   // [1536]
    const float* emb_dec = (const float*)d_in[6];   // [32000,256]
    const float* dec_Wx  = (const float*)d_in[7];   // [256,1536]
    const float* dec_Wh  = (const float*)d_in[8];   // [512,1536]
    const float* dec_b   = (const float*)d_in[9];   // [1536]
    const float* fc_W    = (const float*)d_in[10];  // [512,32000]
    const float* fc_b    = (const float*)d_in[11];  // [32000]
    float* out = (float*)d_out;

    size_t off = 0;
    char* base = (char*)d_ws;
    auto alloc = [&](size_t bytes) -> char* {
        char* p = base + off;
        off += (bytes + 255) & ~(size_t)255;
        return p;
    };
    // transposed bf16 weights: [N, K] row-major
    unsigned short* encWxT = (unsigned short*)alloc((size_t)E_ * H3_ * 2);
    unsigned short* encWhT = (unsigned short*)alloc((size_t)H_ * H3_ * 2);
    unsigned short* decWxT = (unsigned short*)alloc((size_t)E_ * H3_ * 2);
    unsigned short* decWhT = (unsigned short*)alloc((size_t)H_ * H3_ * 2);
    unsigned short* fcWT   = (unsigned short*)alloc((size_t)H_ * V_ * 2);
    unsigned short* Xb  = (unsigned short*)alloc((size_t)B_ * E_ * 2);
    unsigned short* Hb  = (unsigned short*)alloc((size_t)B_ * H_ * 2);
    unsigned short* RHb = (unsigned short*)alloc((size_t)B_ * H_ * 2);
    float* H      = (float*)alloc((size_t)B_ * H_ * 4);
    float* Z      = (float*)alloc((size_t)B_ * H_ * 4);
    float* G      = (float*)alloc((size_t)B_ * H3_ * 4);
    float* G2     = (float*)alloc((size_t)B_ * H3_ * 4);
    float* M3     = (float*)alloc((size_t)B_ * H_ * 4);
    float* logits = (float*)alloc((size_t)B_ * V_ * 4);
    float* ce_buf = (float*)alloc(B_ * 4);
    int*   tok    = (int*)  alloc(B_ * 4);
    float* loss   = (float*)alloc(4);
    (void)ws_size; (void)n_in; (void)in_sizes; (void)out_size;

    // --- per-call: transpose + quantize weights fp32[K,N] -> bf16[N,K] ---
    auto cvtT = [&](const float* s, unsigned short* d, int K, int N) {
        cvt_transpose_bf16<<<dim3(N / TT, K / TT), dim3(TT, 8), 0, stream>>>(
            s, d, K, N);
    };
    cvtT(enc_Wx, encWxT, E_, H3_);
    cvtT(enc_Wh, encWhT, H_, H3_);
    cvtT(dec_Wx, decWxT, E_, H3_);
    cvtT(dec_Wh, decWhT, H_, H3_);
    cvtT(fc_W,   fcWT,   H_, V_);

    init_state<<<(B_ * H_ + 255) / 256, 256, 0, stream>>>(H, Hb, tok, loss);

    // --- encoder: 64 GRU steps ---
    for (int t = 0; t < S_IN; ++t) {
        gather_embed_bf16<<<B_, 64, 0, stream>>>(emb_enc, inp, S_IN, t, Xb, E_);
        wmma_gemm_bf16<<<H3_ / NBLK, 128, 0, stream>>>(Xb, E_, encWxT, E_,
                                                       enc_b, G, H3_, E_);
        wmma_gemm_bf16<<<H3_ / NBLK, 128, 0, stream>>>(Hb, H_, encWhT, H_,
                                                       nullptr, G2, H3_, H_);
        gru_zr<<<B_, H_, 0, stream>>>(G, G2, H, Z, RHb);
        wmma_gemm_bf16<<<H_ / NBLK, 128, 0, stream>>>(
            RHb, H_, encWhT + (size_t)2 * H_ * H_, H_, nullptr, M3, H_, H_);
        gru_hnew<<<B_, H_, 0, stream>>>(G, M3, Z, H, Hb);
    }

    // --- decoder: greedy argmax feedback, 31 steps over targ[:, 1:] ---
    for (int t = 1; t < T_TAR; ++t) {
        gather_embed_bf16<<<B_, 64, 0, stream>>>(emb_dec, tok, 1, 0, Xb, E_);
        wmma_gemm_bf16<<<H3_ / NBLK, 128, 0, stream>>>(Xb, E_, decWxT, E_,
                                                       dec_b, G, H3_, E_);
        wmma_gemm_bf16<<<H3_ / NBLK, 128, 0, stream>>>(Hb, H_, decWhT, H_,
                                                       nullptr, G2, H3_, H_);
        gru_zr<<<B_, H_, 0, stream>>>(G, G2, H, Z, RHb);
        wmma_gemm_bf16<<<H_ / NBLK, 128, 0, stream>>>(
            RHb, H_, decWhT + (size_t)2 * H_ * H_, H_, nullptr, M3, H_, H_);
        gru_hnew<<<B_, H_, 0, stream>>>(G, M3, Z, H, Hb);
        // FC head: [64,512] @ [512,32000]  (fcWT L2-resident, 33 MB)
        wmma_gemm_bf16<<<V_ / NBLK, 128, 0, stream>>>(Hb, H_, fcWT, H_,
                                                      fc_b, logits, V_, H_);
        dec_softmax_loss<<<B_, 256, 0, stream>>>(logits, targ, t, tok, ce_buf);
        accum_loss<<<1, 64, 0, stream>>>(ce_buf, loss);
    }

    write_out<<<1, 1, 0, stream>>>(loss, out);
}
